// GCNRegression_51170240364590
// MI455X (gfx1250) — matrix operand order repaired and checked by reading, not compiled, not run on hardware
//
#include <hip/hip_runtime.h>

#define NN 50000
#define EE 800000
#define HD 128
#define ROWS_PER_BLOCK 80
#define GEMM_BLOCKS (NN / ROWS_PER_BLOCK) /* 625 */
#define EPS 1e-5f

typedef __attribute__((ext_vector_type(16))) __bf16 v16bf;
typedef __attribute__((ext_vector_type(8)))  float  v8f;

struct alignas(16) U4 { unsigned int x, y, z, w; };

static __device__ __forceinline__ unsigned short f2bf(float f) {
  unsigned int u = __builtin_bit_cast(unsigned int, f);
  unsigned int r = 0x7FFFu + ((u >> 16) & 1u);   // round-to-nearest-even
  return (unsigned short)((u + r) >> 16);
}
static __device__ __forceinline__ float bf2f(unsigned short s) {
  unsigned int u = ((unsigned int)s) << 16;
  return __builtin_bit_cast(float, u);
}
// Build a 16-element bf16 fragment from two 16-byte LDS chunks.
static __device__ __forceinline__ v16bf make_frag(const unsigned short* p0,
                                                  const unsigned short* p1) {
  struct alignas(32) Bits { U4 a, b; } bits;
  bits.a = *(const U4*)p0;
  bits.b = *(const U4*)p1;
  return __builtin_bit_cast(v16bf, bits);
}

// ---------------------------------------------------------------- CSR build
__global__ __launch_bounds__(256) void cnt_kernel(const long long* __restrict__ dst,
                                                  int* __restrict__ cnt) {
  int e = blockIdx.x * 256 + threadIdx.x;
  if (e < EE) atomicAdd(&cnt[(int)dst[e]], 1);
}

// Single-block sequential-chunk exclusive scan over NN counts.
__global__ __launch_bounds__(1024) void scan_kernel(const int* __restrict__ cnt,
                                                    int* __restrict__ rowptr,
                                                    int* __restrict__ cursor) {
  __shared__ int tmp[1024];
  __shared__ int carry;
  if (threadIdx.x == 0) carry = 0;
  __syncthreads();
  for (int base = 0; base < NN; base += 1024) {
    int i = base + (int)threadIdx.x;
    int v = (i < NN) ? cnt[i] : 0;
    tmp[threadIdx.x] = v;
    __syncthreads();
#pragma unroll
    for (int ofs = 1; ofs < 1024; ofs <<= 1) {
      int t = (threadIdx.x >= (unsigned)ofs) ? tmp[threadIdx.x - ofs] : 0;
      __syncthreads();
      tmp[threadIdx.x] += t;
      __syncthreads();
    }
    int excl = tmp[threadIdx.x] - v;
    if (i < NN) {
      int rp = carry + excl;
      rowptr[i] = rp;
      cursor[i] = rp;
    }
    __syncthreads();
    if (threadIdx.x == 1023) carry += tmp[1023];
    __syncthreads();
  }
}

__global__ __launch_bounds__(256) void scatter_kernel(const long long* __restrict__ src,
                                                      const long long* __restrict__ dst,
                                                      int* __restrict__ cursor,
                                                      int* __restrict__ csr_src) {
  int e = blockIdx.x * 256 + threadIdx.x;
  if (e < EE) {
    int d = (int)dst[e];
    int pos = atomicAdd(&cursor[d], 1);
    csr_src[pos] = (int)src[e];
  }
}

__global__ __launch_bounds__(256) void dinv_kernel(const int* __restrict__ cnt,
                                                   float* __restrict__ dinv) {
  int i = blockIdx.x * 256 + threadIdx.x;
  if (i < NN) dinv[i] = rsqrtf((float)cnt[i] + 1.0f);  // + self loop; always >= 1
}

// -------------------------------------------------- split W (fp32 -> bf16 hi/lo, transposed [n][k])
__global__ __launch_bounds__(256) void wsplit_kernel(const float* __restrict__ W,
                                                     unsigned short* __restrict__ hiT,
                                                     unsigned short* __restrict__ loT) {
  int idx = blockIdx.x * 256 + threadIdx.x;   // 0..16383 ; W[k][n]
  int k = idx >> 7;
  int n = idx & 127;
  float w = W[idx];
  unsigned short hi = f2bf(w);
  hiT[n * HD + k] = hi;
  loT[n * HD + k] = f2bf(w - bf2f(hi));
}

// -------------------------------------------------- GEMM: Out[N x 128] = A[N x 128] @ W[128 x 128]
// Split-bf16 (hi/lo) x3 WMMA per K-step for ~fp32 accuracy.
__global__ __launch_bounds__(256) void gcn_gemm_wmma(
    const float* __restrict__ A, const unsigned short* __restrict__ WhiT,
    const unsigned short* __restrict__ WloT, float* __restrict__ Out) {
  __shared__ unsigned short sAhi[ROWS_PER_BLOCK * HD];
  __shared__ unsigned short sAlo[ROWS_PER_BLOCK * HD];
  __shared__ unsigned short sBhi[HD * HD];
  __shared__ unsigned short sBlo[HD * HD];

  const int tid  = threadIdx.x;
  const int row0 = blockIdx.x * ROWS_PER_BLOCK;

  // Stage + split the 80x128 fp32 A tile into bf16 hi/lo (40 elems per thread).
#pragma unroll
  for (int j = 0; j < 5; ++j) {
    int base = (tid + j * 256) * 8;           // linear element index
    int r = base >> 7;
    int c = base & 127;
    const float4* ap = (const float4*)(A + (size_t)(row0 + r) * HD + c);
    float4 v0 = ap[0];
    float4 v1 = ap[1];
    float f[8] = {v0.x, v0.y, v0.z, v0.w, v1.x, v1.y, v1.z, v1.w};
#pragma unroll
    for (int i = 0; i < 8; ++i) {
      unsigned short hi = f2bf(f[i]);
      sAhi[base + i] = hi;
      sAlo[base + i] = f2bf(f[i] - bf2f(hi));
    }
  }
  // Stage pre-split transposed W (2 x 32 KB) into LDS with 16B copies.
  {
    const U4* sh = (const U4*)WhiT;
    const U4* sl = (const U4*)WloT;
    U4* dh = (U4*)sBhi;
    U4* dl = (U4*)sBlo;
#pragma unroll
    for (int i = 0; i < 8; ++i) {
      dh[tid + 256 * i] = sh[tid + 256 * i];
      dl[tid + 256 * i] = sl[tid + 256 * i];
    }
  }
  __syncthreads();

  const int wave = tid >> 5;        // 8 waves -> 8 column tiles of 16
  const int lane = tid & 31;
  const int n0   = wave * 16;
  const int m    = lane & 15;
  const int half = lane >> 4;       // lane group selects K half (A/B) and M half (D)
  const int nloc = n0 + m;

  v8f acc[5] = {};

#pragma unroll
  for (int kt = 0; kt < 4; ++kt) {
    const int k0 = kt * 32;
    // B fragment: 16 contiguous K per lane (lanes 0-15: k0..k0+15, lanes 16-31: +16)
    const int kc = k0 + half * 16;
    v16bf bh = make_frag(&sBhi[nloc * HD + kc], &sBhi[nloc * HD + kc + 8]);
    v16bf bl = make_frag(&sBlo[nloc * HD + kc], &sBlo[nloc * HD + kc + 8]);
    // A fragment: 16 contiguous K per lane (matches B)
    const int kb = k0 + half * 16;
#pragma unroll
    for (int rt = 0; rt < 5; ++rt) {
      const int ar = (rt * 16 + m) * HD;
      v16bf ah = make_frag(&sAhi[ar + kb], &sAhi[ar + kb + 8]);
      v16bf al = make_frag(&sAlo[ar + kb], &sAlo[ar + kb + 8]);
      acc[rt] = __builtin_amdgcn_wmma_f32_16x16x32_bf16(false, ah, false, bh,
                                                        (short)0, acc[rt], false, false);
      acc[rt] = __builtin_amdgcn_wmma_f32_16x16x32_bf16(false, al, false, bh,
                                                        (short)0, acc[rt], false, false);
      acc[rt] = __builtin_amdgcn_wmma_f32_16x16x32_bf16(false, ah, false, bl,
                                                        (short)0, acc[rt], false, false);
    }
  }

  // D layout: vgpr r -> row r + 8*half, col = lane&15 within tile.
#pragma unroll
  for (int rt = 0; rt < 5; ++rt) {
#pragma unroll
    for (int r = 0; r < 8; ++r) {
      Out[(size_t)(row0 + rt * 16 + half * 8 + r) * HD + nloc] = acc[rt][r];
    }
  }
}

// -------------------------------------------------- atomic-free aggregation
// One wave per node: agg[n] = dinv[n]^2 * t[n] + sum_{s in N(n)} dinv[n]*dinv[s]*t[s]
__global__ __launch_bounds__(256) void node_agg_kernel(const float* __restrict__ t,
                                                       const int* __restrict__ rowptr,
                                                       const int* __restrict__ cnt,
                                                       const int* __restrict__ csr_src,
                                                       const float* __restrict__ dinv,
                                                       float* __restrict__ agg) {
  int node = blockIdx.x * 8 + (int)(threadIdx.x >> 5);
  if (node >= NN) return;
  int lane = threadIdx.x & 31;
  float di = dinv[node];
  // self-loop term initializes the accumulator (also fully initializes agg)
  float4 acc = ((const float4*)(t + (size_t)node * HD))[lane];
  float w0 = di * di;
  acc.x *= w0; acc.y *= w0; acc.z *= w0; acc.w *= w0;
  int beg = rowptr[node];
  int num = cnt[node];
  for (int j = 0; j < num; ++j) {
    int s = csr_src[beg + j];
    float w = di * dinv[s];
    float4 v = ((const float4*)(t + (size_t)s * HD))[lane];
    acc.x += v.x * w; acc.y += v.y * w; acc.z += v.z * w; acc.w += v.w * w;
  }
  ((float4*)(agg + (size_t)node * HD))[lane] = acc;
}

// -------------------------------------------------- bias + BN(eval) + ReLU
__global__ __launch_bounds__(256) void bn_relu_kernel(const float* __restrict__ agg,
                                                      const float* __restrict__ b,
                                                      const float* __restrict__ g,
                                                      const float* __restrict__ be,
                                                      const float* __restrict__ m,
                                                      const float* __restrict__ v,
                                                      float* __restrict__ out) {
  int idx = blockIdx.x * 256 + threadIdx.x;   // one float4 per thread; NN*32 total
  if (idx >= NN * 32) return;
  int c0 = (idx & 31) * 4;
  float4 x = ((const float4*)agg)[idx];
  float xs[4] = {x.x, x.y, x.z, x.w};
  float rs[4];
#pragma unroll
  for (int i = 0; i < 4; ++i) {
    int c = c0 + i;
    float s = g[c] * rsqrtf(v[c] + EPS);
    float r = (xs[i] + b[c] - m[c]) * s + be[c];
    rs[i] = fmaxf(r, 0.0f);
  }
  float4 o = {rs[0], rs[1], rs[2], rs[3]};
  ((float4*)out)[idx] = o;
}

// -------------------------------------------------- layer 3: GEMV (128 -> 1)
__global__ __launch_bounds__(256) void gemv_kernel(const float* __restrict__ h,
                                                   const float* __restrict__ W3,
                                                   float* __restrict__ outp) {
  __shared__ float w3s[HD];
  if (threadIdx.x < HD) w3s[threadIdx.x] = W3[threadIdx.x];
  __syncthreads();
  int row = blockIdx.x * 256 + threadIdx.x;
  if (row >= NN) return;
  const float4* hp = (const float4*)(h + (size_t)row * HD);
  float sum = 0.0f;
#pragma unroll
  for (int i = 0; i < 32; ++i) {
    float4 t = hp[i];
    sum += t.x * w3s[4 * i] + t.y * w3s[4 * i + 1] +
           t.z * w3s[4 * i + 2] + t.w * w3s[4 * i + 3];
  }
  outp[row] = sum;
}

// Final scalar aggregation via CSR (atomic-free), + b3.
__global__ __launch_bounds__(256) void out_node_kernel(const float* __restrict__ pre,
                                                       const int* __restrict__ rowptr,
                                                       const int* __restrict__ cnt,
                                                       const int* __restrict__ csr_src,
                                                       const float* __restrict__ dinv,
                                                       const float* __restrict__ b3,
                                                       float* __restrict__ out) {
  int i = blockIdx.x * 256 + threadIdx.x;
  if (i >= NN) return;
  float di = dinv[i];
  float sum = pre[i] * di * di;
  int beg = rowptr[i];
  int num = cnt[i];
  for (int j = 0; j < num; ++j) {
    int s = csr_src[beg + j];
    sum += pre[s] * dinv[s] * di;
  }
  out[i] = sum + b3[0];
}

// ----------------------------------------------------------------------------
extern "C" void kernel_launch(void* const* d_in, const int* in_sizes, int n_in,
                              void* d_out, int out_size, void* d_ws, size_t ws_size,
                              hipStream_t stream) {
  const float*     x  = (const float*)d_in[0];
  const long long* ei = (const long long*)d_in[1];
  const float* W1 = (const float*)d_in[2];
  const float* b1 = (const float*)d_in[3];
  const float* g1 = (const float*)d_in[4];
  const float* be1 = (const float*)d_in[5];
  const float* m1 = (const float*)d_in[6];
  const float* v1 = (const float*)d_in[7];
  const float* W2 = (const float*)d_in[8];
  const float* b2 = (const float*)d_in[9];
  const float* g2 = (const float*)d_in[10];
  const float* be2 = (const float*)d_in[11];
  const float* m2 = (const float*)d_in[12];
  const float* v2 = (const float*)d_in[13];
  const float* W3 = (const float*)d_in[14];
  const float* b3 = (const float*)d_in[15];
  float* out = (float*)d_out;

  const long long* srcIdx = ei;        // edge_index[0]
  const long long* dstIdx = ei + EE;   // edge_index[1]

  char* ws = (char*)d_ws;
  size_t off = 0;
  float* dinv   = (float*)(ws + off); off += 262144;   // N floats
  int*   cnt    = (int*)(ws + off);   off += 262144;   // N ints (degree w/o self loop)
  int*   rowptr = (int*)(ws + off);   off += 262144;   // N ints
  int*   cursor = (int*)(ws + off);   off += 262144;   // N ints
  int*   csr    = (int*)(ws + off);   off += 3276800;  // E ints (src grouped by dst)
  float* preb   = (float*)(ws + off); off += 262144;   // N floats (layer-3 pre-agg)
  float* bufA   = (float*)(ws + off); off += (size_t)NN * HD * 4;   // 25.6 MB
  float* bufB   = (float*)(ws + off); off += (size_t)NN * HD * 4;
  unsigned short* w1hi = (unsigned short*)(ws + off); off += (size_t)HD * HD * 2;
  unsigned short* w1lo = (unsigned short*)(ws + off); off += (size_t)HD * HD * 2;
  unsigned short* w2hi = (unsigned short*)(ws + off); off += (size_t)HD * HD * 2;
  unsigned short* w2lo = (unsigned short*)(ws + off); off += (size_t)HD * HD * 2;

  const int ELEM_BLKS = (NN * 32) / 256;        // 6250 (exact)
  const int NODE_BLKS = (NN + 255) / 256;       // 196
  const int EDGE_BLKS = (EE + 255) / 256;       // 3125 (exact)
  const int WAVE_BLKS = (NN + 7) / 8;           // 6250 (one wave per node)

  // ---- CSR build + normalization coefficients (once; reused by all layers)
  hipMemsetAsync(cnt, 0, NN * sizeof(int), stream);
  cnt_kernel<<<EDGE_BLKS, 256, 0, stream>>>(dstIdx, cnt);
  scan_kernel<<<1, 1024, 0, stream>>>(cnt, rowptr, cursor);
  dinv_kernel<<<NODE_BLKS, 256, 0, stream>>>(cnt, dinv);
  scatter_kernel<<<EDGE_BLKS, 256, 0, stream>>>(srcIdx, dstIdx, cursor, csr);

  // ---- pre-split weights (tiny)
  wsplit_kernel<<<64, 256, 0, stream>>>(W1, w1hi, w1lo);
  wsplit_kernel<<<64, 256, 0, stream>>>(W2, w2hi, w2lo);

  // ---- layer 1
  gcn_gemm_wmma<<<GEMM_BLOCKS, 256, 0, stream>>>(x, w1hi, w1lo, bufA);
  node_agg_kernel<<<WAVE_BLKS, 256, 0, stream>>>(bufA, rowptr, cnt, csr, dinv, bufB);
  bn_relu_kernel<<<ELEM_BLKS, 256, 0, stream>>>(bufB, b1, g1, be1, m1, v1, bufA);

  // ---- layer 2
  gcn_gemm_wmma<<<GEMM_BLOCKS, 256, 0, stream>>>(bufA, w2hi, w2lo, bufB);
  node_agg_kernel<<<WAVE_BLKS, 256, 0, stream>>>(bufB, rowptr, cnt, csr, dinv, bufA);
  bn_relu_kernel<<<ELEM_BLKS, 256, 0, stream>>>(bufA, b2, g2, be2, m2, v2, bufB);

  // ---- layer 3 (GEMV + CSR scalar aggregation)
  gemv_kernel<<<NODE_BLKS, 256, 0, stream>>>(bufB, W3, preb);
  out_node_kernel<<<NODE_BLKS, 256, 0, stream>>>(preb, rowptr, cnt, csr, dinv, b3, out);
}
